// KernelNN_82970178224518
// MI455X (gfx1250) — matrix operand order repaired and checked by reading, not compiled, not run on hardware
//
#include <hip/hip_runtime.h>
#include <hip/hip_bf16.h>

// ---------------------------------------------------------------------------
// NNConv message-passing GNN for MI455X (gfx1250, wave32, WMMA).
//
// Stages:
//   P1) pack k2_w / k3_w (f32 row-major) -> f16 WMMA B-fragment layout
//   L1) h1 = relu(edge_attr @ k1_w + b1)            [E,256] f16   (VALU, K=6)
//   L2) h2 = relu(h1 @ k2_w + b2)                   [E,256] f16   (WMMA, NT=4)
//   L3) W_e = h2 @ k3_w + b3, stored TRANSPOSED     [E,32out,32in] f16 (WMMA, NT=8)
//   I ) h0 = x*fc1_w + fc1_b ; agg=0 ; deg via atomics
//   4x) msg  : per-edge GEMV (h_src . W_e) -> atomicAdd scatter into agg
//       comb : h = relu(agg/deg + h @ root_w + conv_b)
//   F ) out = h @ fc2_w + fc2_b   (wave reduction)
// ---------------------------------------------------------------------------

#define NNODES 10000
#define NEDGES 160000
#define WIDTH  32
#define KERW   256
#define KERIN  6
#define DEPTH  4

typedef _Float16 v8h  __attribute__((ext_vector_type(8)));
typedef _Float16 v16h __attribute__((ext_vector_type(16)));
typedef float    v8f  __attribute__((ext_vector_type(8)));

// ---------------------------------------------------------------------------
// Pack B (f32 row-major [K,N]) into f16 WMMA 32x16 B-fragment blocks.
// Block (kb, nb): for lane l, half h:
//   K = kb*32 + (h&7) + ((h&8)<<1) + ((l&16)>>1),  col = nb*16 + (l&15)
// Dest layout: Bp[((nb*KB + kb)*32 + lane)*16 + h]  (32B contiguous per lane)
// ---------------------------------------------------------------------------
__global__ void pack_b_kernel(const float* __restrict__ B, _Float16* __restrict__ Bp,
                              int K, int N) {
    size_t idx = (size_t)blockIdx.x * 256 + threadIdx.x;
    size_t total = (size_t)K * N;
    if (idx >= total) return;
    int h    = (int)(idx & 15);
    int lane = (int)((idx >> 4) & 31);
    size_t blk = idx >> 9;
    int KB = K >> 5;
    int kb = (int)(blk % KB);
    int nb = (int)(blk / KB);
    int row = kb * 32 + (h & 7) + ((h & 8) << 1) + ((lane & 16) >> 1);
    int col = nb * 16 + (lane & 15);
    Bp[idx] = (_Float16)B[(size_t)row * N + col];
}

// ---------------------------------------------------------------------------
// Layer 1: h1[e][c] = relu(sum_k ea[e][k]*k1w[k][c] + b1[c]) ; f16 row-major.
// blockIdx.x = edge (uniform -> ea loads become scalar loads), tid = channel.
// ---------------------------------------------------------------------------
__global__ __launch_bounds__(KERW) void layer1_kernel(
    const float* __restrict__ ea, const float* __restrict__ k1w,
    const float* __restrict__ k1b, _Float16* __restrict__ h1) {
    int e = blockIdx.x;
    int c = threadIdx.x;
    float acc = k1b[c];
#pragma unroll
    for (int k = 0; k < KERIN; ++k)
        acc = fmaf(ea[(size_t)e * KERIN + k], k1w[k * KERW + c], acc);
    h1[(size_t)e * KERW + c] = (_Float16)fmaxf(acc, 0.f);
}

// ---------------------------------------------------------------------------
// WMMA GEMM: C = act(A[M,K] @ B[K,N] + bias), A f16 row-major, B pre-packed,
// C f16. 256 threads = 8 waves; block tile 128(M) x NT*16(N); each wave owns
// a 16 x NT*16 strip = NT accumulators, so each A fragment feeds NT WMMAs.
// TRANSOUT!=0 stores W_e transposed per edge: n = i*32+o -> C[e*1024 + o*32 + i].
// ---------------------------------------------------------------------------
template <int NT, int RELU, int TRANSOUT>
__global__ __launch_bounds__(256) void gemm_wmma_kernel(
    const _Float16* __restrict__ A, const _Float16* __restrict__ Bp,
    const float* __restrict__ bias, _Float16* __restrict__ C,
    int M, int N, int K) {
    const int KB   = K >> 5;
    int lane   = threadIdx.x & 31;
    int wave   = threadIdx.x >> 5;
    int m_base = blockIdx.x * 128 + wave * 16;
    int n_base = blockIdx.y * (NT * 16);

    int arow_i = m_base + (lane & 15);
    if (arow_i >= M) arow_i = M - 1;          // clamp (stores are guarded)
    const _Float16* arow = A + (size_t)arow_i * K;
    int ksel = (lane & 16) ? 8 : 0;
    int nb0  = n_base >> 4;

    v8f acc[NT];
#pragma unroll
    for (int t = 0; t < NT; ++t) acc[t] = v8f{};

    for (int kb = 0; kb < KB; ++kb) {
        int k0 = kb * 32 + ksel;
        if (kb + 1 < KB)
            __builtin_prefetch(arow + k0 + 32, 0, 1);   // global_prefetch_b8
        v8h alo = *(const v8h*)(arow + k0);
        v8h ahi = *(const v8h*)(arow + k0 + 16);
        v16h a;
#pragma unroll
        for (int j = 0; j < 8; ++j) { a[j] = alo[j]; a[j + 8] = ahi[j]; }

#pragma unroll
        for (int t = 0; t < NT; ++t) {
            const _Float16* bptr =
                Bp + (((size_t)(nb0 + t) * KB + kb) * 32 + lane) * 16;
            v8h blo = *(const v8h*)(bptr);
            v8h bhi = *(const v8h*)(bptr + 8);
            v16h b;
#pragma unroll
            for (int j = 0; j < 8; ++j) { b[j] = blo[j]; b[j + 8] = bhi[j]; }
            acc[t] = __builtin_amdgcn_wmma_f32_16x16x32_f16(
                false, a, false, b, (short)0, acc[t], false, false);
        }
    }

    int m_lo = (lane & 16) ? 8 : 0;
#pragma unroll
    for (int t = 0; t < NT; ++t) {
#pragma unroll
        for (int v = 0; v < 8; ++v) {
            int r   = m_base + m_lo + v;
            int cgl = n_base + t * 16 + (lane & 15);
            if (r < M) {
                float val = acc[t][v] + bias[cgl];
                if (RELU) val = fmaxf(val, 0.f);
                size_t di;
                if (TRANSOUT) {
                    int i = cgl >> 5, o = cgl & 31;      // n = i*32 + o
                    di = (size_t)r * (WIDTH * WIDTH) + (size_t)o * WIDTH + i;
                } else {
                    di = (size_t)r * N + cgl;
                }
                C[di] = (_Float16)val;
            }
        }
    }
}

// ---------------------------------------------------------------------------
// Init: h0 = x*fc1_w + fc1_b ; agg = 0 ; deg = 0. One thread per (node, ch).
// ---------------------------------------------------------------------------
__global__ void init_kernel(const float* __restrict__ x,
                            const float* __restrict__ fc1w,
                            const float* __restrict__ fc1b,
                            float* __restrict__ h, float* __restrict__ agg,
                            float* __restrict__ deg, int Nn) {
    int idx = blockIdx.x * blockDim.x + threadIdx.x;
    if (idx >= Nn * WIDTH) return;
    int node = idx >> 5, j = idx & 31;
    h[idx]   = fmaf(x[node], fc1w[j], fc1b[j]);
    agg[idx] = 0.f;
    if (j == 0) deg[node] = 0.f;
}

__global__ void degree_kernel(const int* __restrict__ dst, float* __restrict__ deg,
                              int E) {
    int e = blockIdx.x * blockDim.x + threadIdx.x;
    if (e < E) unsafeAtomicAdd(&deg[dst[e]], 1.0f);
}

// ---------------------------------------------------------------------------
// Message: one wave per edge. lane = output channel o.
//   msg_o = sum_i h[src][i] * Wt[e][o][i]   (Wt stored [out][in], 64B/lane)
// Scatter: atomicAdd into agg[dst].
// ---------------------------------------------------------------------------
__global__ __launch_bounds__(256) void msg_kernel(
    const float* __restrict__ h, const _Float16* __restrict__ Wt,
    const int* __restrict__ src, const int* __restrict__ dst,
    float* __restrict__ agg, int E) {
    int wid  = (blockIdx.x * blockDim.x + threadIdx.x) >> 5;
    int lane = threadIdx.x & 31;
    if (wid >= E) return;
    int s = src[wid], d = dst[wid];
    float hv = h[(size_t)s * WIDTH + lane];
    const v8h* wp = (const v8h*)(Wt + (size_t)wid * (WIDTH * WIDTH) + lane * WIDTH);
    float acc = 0.f;
#pragma unroll
    for (int c = 0; c < 4; ++c) {
        v8h w = wp[c];
#pragma unroll
        for (int j = 0; j < 8; ++j) {
            float hi = __shfl(hv, c * 8 + j, 32);
            acc = fmaf((float)w[j], hi, acc);
        }
    }
    unsafeAtomicAdd(&agg[(size_t)d * WIDTH + lane], acc);
}

// ---------------------------------------------------------------------------
// Combine: one wave per node. h = relu(agg/deg + h@root_w + conv_b);
// also zeroes agg for the next iteration.
// ---------------------------------------------------------------------------
__global__ __launch_bounds__(256) void combine_kernel(
    float* __restrict__ h, float* __restrict__ agg,
    const float* __restrict__ deg, const float* __restrict__ rootw,
    const float* __restrict__ convb, int Nn) {
    int wid  = (blockIdx.x * blockDim.x + threadIdx.x) >> 5;
    int lane = threadIdx.x & 31;
    if (wid >= Nn) return;
    size_t base = (size_t)wid * WIDTH;
    float hv = h[base + lane];
    float a  = agg[base + lane];
    agg[base + lane] = 0.f;
    float invd = 1.0f / fmaxf(deg[wid], 1.0f);
    float acc = fmaf(a, invd, convb[lane]);
#pragma unroll
    for (int i = 0; i < WIDTH; ++i)
        acc = fmaf(__shfl(hv, i, 32), rootw[i * WIDTH + lane], acc);
    h[base + lane] = fmaxf(acc, 0.f);
}

// ---------------------------------------------------------------------------
// Final: out[n] = h[n] . fc2_w + fc2_b  (wave32 shfl_xor reduction)
// ---------------------------------------------------------------------------
__global__ __launch_bounds__(256) void final_kernel(
    const float* __restrict__ h, const float* __restrict__ fc2w,
    const float* __restrict__ fc2b, float* __restrict__ out, int Nn) {
    int wid  = (blockIdx.x * blockDim.x + threadIdx.x) >> 5;
    int lane = threadIdx.x & 31;
    if (wid >= Nn) return;
    float p = h[(size_t)wid * WIDTH + lane] * fc2w[lane];
#pragma unroll
    for (int off = 16; off > 0; off >>= 1) p += __shfl_xor(p, off, 32);
    if (lane == 0) out[wid] = p + fc2b[0];
}

// ---------------------------------------------------------------------------
// Host-side launch
// ---------------------------------------------------------------------------
static inline size_t align256(size_t v) { return (v + 255) & ~(size_t)255; }

extern "C" void kernel_launch(void* const* d_in, const int* in_sizes, int n_in,
                              void* d_out, int out_size, void* d_ws, size_t ws_size,
                              hipStream_t stream) {
    const float* x       = (const float*)d_in[0];
    const int*   eidx    = (const int*)d_in[1];
    const float* eattr   = (const float*)d_in[2];
    const float* fc1_w   = (const float*)d_in[3];
    const float* fc1_b   = (const float*)d_in[4];
    const float* k1_w    = (const float*)d_in[5];
    const float* k1_b    = (const float*)d_in[6];
    const float* k2_w    = (const float*)d_in[7];
    const float* k2_b    = (const float*)d_in[8];
    const float* k3_w    = (const float*)d_in[9];
    const float* k3_b    = (const float*)d_in[10];
    const float* root_w  = (const float*)d_in[11];
    const float* conv_b  = (const float*)d_in[12];
    const float* fc2_w   = (const float*)d_in[13];
    const float* fc2_b   = (const float*)d_in[14];

    const int* src = eidx;              // edge_index[0]
    const int* dst = eidx + NEDGES;     // edge_index[1]

    // Workspace partition
    char* ws = (char*)d_ws;
    size_t off = 0;
    _Float16* h1  = (_Float16*)(ws + off); off = align256(off + (size_t)NEDGES * KERW * 2);
    _Float16* h2  = (_Float16*)(ws + off); off = align256(off + (size_t)NEDGES * KERW * 2);
    _Float16* We  = (_Float16*)(ws + off); off = align256(off + (size_t)NEDGES * WIDTH * WIDTH * 2);
    _Float16* k2p = (_Float16*)(ws + off); off = align256(off + (size_t)KERW * KERW * 2);
    _Float16* k3p = (_Float16*)(ws + off); off = align256(off + (size_t)KERW * WIDTH * WIDTH * 2);
    float*    hbuf= (float*)(ws + off);   off = align256(off + (size_t)NNODES * WIDTH * 4);
    float*    agg = (float*)(ws + off);   off = align256(off + (size_t)NNODES * WIDTH * 4);
    float*    deg = (float*)(ws + off);   off = align256(off + (size_t)NNODES * 4);
    (void)ws_size; (void)n_in; (void)in_sizes; (void)out_size;

    // P1: pack weights into WMMA B-fragment layout
    pack_b_kernel<<<(KERW * KERW + 255) / 256, 256, 0, stream>>>(k2_w, k2p, KERW, KERW);
    pack_b_kernel<<<(KERW * WIDTH * WIDTH + 255) / 256, 256, 0, stream>>>(k3_w, k3p, KERW, WIDTH * WIDTH);

    // L1: edge feature expansion
    layer1_kernel<<<NEDGES, KERW, 0, stream>>>(eattr, k1_w, k1_b, h1);

    // L2: h2 = relu(h1 @ k2)   [E,256]  (wave strip 16x64)
    {
        dim3 grid((NEDGES + 127) / 128, KERW / 64);
        gemm_wmma_kernel<4, 1, 0><<<grid, 256, 0, stream>>>(h1, k2p, k2_b, h2,
                                                            NEDGES, KERW, KERW);
    }
    // L3: W_e = h2 @ k3  (stored transposed per edge)  [E,1024] (wave strip 16x128)
    {
        dim3 grid((NEDGES + 127) / 128, (WIDTH * WIDTH) / 128);
        gemm_wmma_kernel<8, 0, 1><<<grid, 256, 0, stream>>>(h2, k3p, k3_b, We,
                                                            NEDGES, WIDTH * WIDTH, KERW);
    }

    // Init node state + degrees
    init_kernel<<<(NNODES * WIDTH + 255) / 256, 256, 0, stream>>>(x, fc1_w, fc1_b,
                                                                  hbuf, agg, deg, NNODES);
    degree_kernel<<<(NEDGES + 255) / 256, 256, 0, stream>>>(dst, deg, NEDGES);

    // Message passing iterations
    for (int it = 0; it < DEPTH; ++it) {
        msg_kernel<<<(NEDGES * 32 + 255) / 256, 256, 0, stream>>>(hbuf, We, src, dst,
                                                                  agg, NEDGES);
        combine_kernel<<<(NNODES * 32 + 255) / 256, 256, 0, stream>>>(hbuf, agg, deg,
                                                                      root_w, conv_b, NNODES);
    }

    // Readout
    final_kernel<<<(NNODES * 32 + 255) / 256, 256, 0, stream>>>(hbuf, fc2_w, fc2_b,
                                                                (float*)d_out, NNODES);
}
